// transformer_block_5265629905478
// MI455X (gfx1250) — matrix operand order repaired and checked by reading, not compiled, not run on hardware
//
#include <hip/hip_runtime.h>
#include <math.h>

#define N_NODES 65536
#define DIM     1280
#define DIM2    2560
#define NHEAD   2
#define NGRAPH  1024
#define INV_SCALE 0.0279508497187474f   // 1/sqrt(1280)
#define LN_EPS  1e-5f

typedef unsigned short bfu;
typedef __attribute__((ext_vector_type(8)))  unsigned short v8us;
typedef __attribute__((ext_vector_type(16))) unsigned short v16us;
typedef __attribute__((ext_vector_type(16))) __bf16        v16bf;
typedef __attribute__((ext_vector_type(8)))  float         v8f;

__device__ __forceinline__ bfu f2bf(float f) {
  union { float f; unsigned u; } x; x.f = f;
  unsigned r = x.u + 0x7FFFu + ((x.u >> 16) & 1u);   // round-to-nearest-even
  return (bfu)(r >> 16);
}
__device__ __forceinline__ float bf2f(bfu v) {
  union { unsigned u; float f; } x; x.u = ((unsigned)v) << 16;
  return x.f;
}

// ---------------------------------------------------------------------------
// Async memory->LDS panel copy (CDNA5 GLOBAL_LOAD_ASYNC_TO_LDS_B128, ASYNCcnt)
// Flat shared pointers carry the LDS byte offset in their low 32 bits.
// ---------------------------------------------------------------------------
__device__ __forceinline__ void async_panel_load(void* lds_dst, const void* gsrc,
                                                 int bytes, int tid, int nthreads)
{
  unsigned loff = (unsigned)(size_t)lds_dst;
  const char* g = (const char*)gsrc;
  for (int i = tid * 16; i < bytes; i += nthreads * 16) {
    asm volatile("global_load_async_to_lds_b128 %0, %1, off"
                 :: "v"(loff + (unsigned)i), "v"(g + i) : "memory");
  }
  asm volatile("s_wait_asynccnt 0x0" ::: "memory");
}

// ---------------------------------------------------------------------------
// GEMM micro-panel, load-all-then-MMA form. The sched_barrier(0) pins the
// whole B-load burst ahead of the WMMA burst so the waitcnt pass emits
// decreasing s_wait_loadcnt values (only the first pair's latency per k-step
// is exposed) instead of the scheduler sinking each load to its use with
// s_wait_loadcnt 0.
// A panel: LDS bf16 row-major (caller passes row-group base), ISA 16-bit A
// 16x32 lane layout. B: global bf16 [out,in] row-major; col tile t of
// col-group cg covers output columns [(cg+8t)*16, +16) => B offset t*128*ldb
// folds into the 24-bit instruction immediate.
// ---------------------------------------------------------------------------
template<int NT>
__device__ __forceinline__ void wmma_panel(const bfu* Ap, int lda, int ksteps,
                                           const bfu* B, long ldb,
                                           int cg, int lane, v8f* acc)
{
  const bfu* abase = Ap + (lane & 15) * lda + ((lane >> 4) << 3);
  const bfu* brow0 = B + (long)((cg << 4) + (lane & 15)) * ldb + ((lane >> 4) << 3);
  for (int kt = 0; kt < ksteps; ++kt) {
    const bfu* ap = abase + kt * 32;
    v8us lo = *(const v8us*)ap;          // K  0..7  (or 8..15 for hi half-wave)
    v8us hi = *(const v8us*)(ap + 16);   // K 16..23 (or 24..31)
    v16us bu[NT];
#pragma unroll
    for (int t = 0; t < NT; ++t)
      bu[t] = __builtin_shufflevector(*(const v8us*)(brow0 + (long)t * 128 * ldb + (long)kt * 32), *(const v8us*)(brow0 + (long)t * 128 * ldb + (long)kt * 32 + 16), 0,1,2,3,4,5,6,7,8,9,10,11,12,13,14,15);
    __builtin_amdgcn_sched_barrier(0);   // keep the load burst ahead of the MMAs
    v16us au = __builtin_shufflevector(lo, hi, 0,1,2,3,4,5,6,7,8,9,10,11,12,13,14,15);
    v16bf a = __builtin_bit_cast(v16bf, au);
#pragma unroll
    for (int t = 0; t < NT; ++t)
      acc[t] = __builtin_amdgcn_wmma_f32_16x16x32_bf16(
          false, a, false, __builtin_bit_cast(v16bf, bu[t]), (short)0,
          acc[t], false, false);
  }
}

// ---------------------------------------------------------------------------
// fp32 -> bf16 conversion
// ---------------------------------------------------------------------------
__global__ __launch_bounds__(256) void k_tobf(const float* __restrict__ src,
                                              bfu* __restrict__ dst, long n)
{
  long i = ((long)blockIdx.x * 256 + threadIdx.x) * 4;
  if (i + 3 < n) {
    float4 v = *(const float4*)(src + i);
    dst[i]   = f2bf(v.x); dst[i+1] = f2bf(v.y);
    dst[i+2] = f2bf(v.z); dst[i+3] = f2bf(v.w);
  } else {
    for (; i < n; ++i) dst[i] = f2bf(src[i]);
  }
}

// ---------------------------------------------------------------------------
// Mt[h][e][d] = sum_c Wk[h][c][e] * Wq[h][c][d]   ([out,in] layout for B use)
// ---------------------------------------------------------------------------
__global__ __launch_bounds__(256) void k_mt(const float* __restrict__ Wq,
                                            const float* __restrict__ Wk,
                                            bfu* __restrict__ Mt)
{
  __shared__ float sK[64][17];
  __shared__ float sQ[64][17];
  const int h  = blockIdx.z;
  const int e0 = blockIdx.y * 16, d0 = blockIdx.x * 16;
  const int tid = threadIdx.x;
  const int el = tid >> 4, dl = tid & 15;
  const long base = (long)h * DIM * DIM;
  float sum = 0.f;
  for (int c0 = 0; c0 < DIM; c0 += 64) {
    for (int s = 0; s < 4; ++s) {
      int i = tid + s * 256;
      int cc = i >> 4, ee = i & 15;
      sK[cc][ee] = Wk[base + (long)(c0 + cc) * DIM + e0 + ee];
      sQ[cc][ee] = Wq[base + (long)(c0 + cc) * DIM + d0 + ee];
    }
    __syncthreads();
    for (int cc = 0; cc < 64; ++cc)
      sum += sK[cc][el] * sQ[cc][dl];
    __syncthreads();
  }
  Mt[base + (long)(e0 + el) * DIM + d0 + dl] = f2bf(sum);
}

// ---------------------------------------------------------------------------
// Cbig[o][h*D + d] = sum_c Wv[h][c][d] * Wc[o][h*D + c]
// Head-interleaved [out, 2560] layout so k_multi runs ONE K=2560 GEMM.
// ---------------------------------------------------------------------------
__global__ __launch_bounds__(256) void k_ct(const float* __restrict__ Wv,
                                            const float* __restrict__ Wc,
                                            bfu* __restrict__ Cbig)
{
  __shared__ float sV[64][17];
  __shared__ float sC[64][17];
  const int h  = blockIdx.z;
  const int o0 = blockIdx.y * 16, d0 = blockIdx.x * 16;
  const int tid = threadIdx.x;
  const int ol = tid >> 4, dl = tid & 15;
  const long baseV = (long)h * DIM * DIM;
  float sum = 0.f;
  for (int c0 = 0; c0 < DIM; c0 += 64) {
    for (int s = 0; s < 4; ++s) {
      int i = tid + s * 256;
      int cc = i >> 4, dd = i & 15;
      sV[cc][dd] = Wv[baseV + (long)(c0 + cc) * DIM + d0 + dd];
      int oo = i >> 6, c2 = i & 63;
      sC[c2][oo] = Wc[(long)(o0 + oo) * (NHEAD * DIM) + h * DIM + c0 + c2];
    }
    __syncthreads();
    for (int cc = 0; cc < 64; ++cc)
      sum += sV[cc][dl] * sC[cc][ol];
    __syncthreads();
  }
  Cbig[(long)(o0 + ol) * DIM2 + h * DIM + d0 + dl] = f2bf(sum);
}

// ---------------------------------------------------------------------------
// att[n,h] = (x_n^T M_h x_n)/sqrt(D)  via  Y = X*M_h ; att = rowsum(Y .* X)
// 512 threads = 16 waves: 2 row-groups x 8 col-groups over a 32-row panel.
// ---------------------------------------------------------------------------
__global__ __launch_bounds__(512, 1) void k_att(const bfu* __restrict__ Xbf,
                                                const bfu* __restrict__ Mt,
                                                float* __restrict__ att)
{
  extern __shared__ char smem[];
  bfu*   Ap    = (bfu*)smem;                     // 32 x 1280 bf16  (81920 B)
  float* att_s = (float*)(smem + 32 * DIM * 2);  // 64 floats
  const int tid = threadIdx.x;
  const long r0 = (long)blockIdx.x * 32;
  async_panel_load(Ap, Xbf + r0 * DIM, 32 * DIM * 2, tid, 512);
  if (tid < 64) att_s[tid] = 0.f;
  __syncthreads();
  const int wave = tid >> 5, lane = tid & 31;
  const int rg = wave >> 3, cg = wave & 7;
  const int mbase = (lane < 16) ? 0 : 8;   // C-layout: lanes 0-15 -> M=j, 16-31 -> M=j+8
  const bfu* Arg = Ap + rg * 16 * DIM;
  for (int h = 0; h < NHEAD; ++h) {
    v8f acc[10] = {};
    wmma_panel<10>(Arg, DIM, DIM / 32, Mt + (long)h * DIM * DIM, DIM, cg, lane, acc);
    float part[8];
#pragma unroll
    for (int j = 0; j < 8; ++j) part[j] = 0.f;
#pragma unroll
    for (int t = 0; t < 10; ++t) {
      int nn = ((cg + (t << 3)) << 4) + (lane & 15);
#pragma unroll
      for (int j = 0; j < 8; ++j)
        part[j] += acc[t][j] * bf2f(Arg[(mbase + j) * DIM + nn]);
    }
#pragma unroll
    for (int j = 0; j < 8; ++j)
      atomicAdd(&att_s[(rg * 16 + mbase + j) * 2 + h], part[j]);
  }
  __syncthreads();
  if (tid < 64)
    att[(r0 + (tid >> 1)) * 2 + (tid & 1)] = att_s[tid] * INV_SCALE;
}

// ---------------------------------------------------------------------------
// Per-graph softmax stats (seg_ids sorted): binary-search bounds, max & expsum
// ---------------------------------------------------------------------------
__global__ __launch_bounds__(256) void k_segstats(const float* __restrict__ att,
                                                  const int* __restrict__ seg,
                                                  float* __restrict__ mx,
                                                  float* __restrict__ dn)
{
  __shared__ int bounds[2];
  __shared__ float red0[256], red1[256];
  const int g = blockIdx.x, tid = threadIdx.x;
  if (tid == 0) {
    int lo = 0, hi = N_NODES;
    while (lo < hi) { int mid = (lo + hi) >> 1; if (seg[mid] < g) lo = mid + 1; else hi = mid; }
    bounds[0] = lo;
    hi = N_NODES;
    while (lo < hi) { int mid = (lo + hi) >> 1; if (seg[mid] <= g) lo = mid + 1; else hi = mid; }
    bounds[1] = lo;
  }
  __syncthreads();
  const int lo = bounds[0], hi = bounds[1];
  float m0 = -3.0e38f, m1 = -3.0e38f;
  for (int i = lo + tid; i < hi; i += 256) {
    m0 = fmaxf(m0, att[i * 2]); m1 = fmaxf(m1, att[i * 2 + 1]);
  }
  red0[tid] = m0; red1[tid] = m1;
  __syncthreads();
  for (int off = 128; off; off >>= 1) {
    if (tid < off) {
      red0[tid] = fmaxf(red0[tid], red0[tid + off]);
      red1[tid] = fmaxf(red1[tid], red1[tid + off]);
    }
    __syncthreads();
  }
  m0 = red0[0]; m1 = red1[0];
  __syncthreads();
  float s0 = 0.f, s1 = 0.f;
  for (int i = lo + tid; i < hi; i += 256) {
    s0 += __expf(att[i * 2] - m0); s1 += __expf(att[i * 2 + 1] - m1);
  }
  red0[tid] = s0; red1[tid] = s1;
  __syncthreads();
  for (int off = 128; off; off >>= 1) {
    if (tid < off) { red0[tid] += red0[tid + off]; red1[tid] += red1[tid + off]; }
    __syncthreads();
  }
  if (tid == 0) {
    mx[g * 2] = m0; mx[g * 2 + 1] = m1;
    dn[g * 2]     = fmaxf(red0[0], 1e-30f);
    dn[g * 2 + 1] = fmaxf(red1[0], 1e-30f);
  }
}

// ---------------------------------------------------------------------------
// In-register LayerNorm epilogue: per-row sums via half-wave shfl_xor
// reductions + LDS atomics; stats land in s1/s2 (32 rows).
// ---------------------------------------------------------------------------
__device__ __forceinline__ void ln_accumulate(const v8f* val, int nt,
                                              float* s1, float* s2,
                                              int rg, int mbase, int lane)
{
  float ps[8], ps2[8];
#pragma unroll
  for (int j = 0; j < 8; ++j) { ps[j] = 0.f; ps2[j] = 0.f; }
  for (int t = 0; t < nt; ++t)
#pragma unroll
    for (int j = 0; j < 8; ++j) { float v = val[t][j]; ps[j] += v; ps2[j] += v * v; }
#pragma unroll
  for (int j = 0; j < 8; ++j) {
#pragma unroll
    for (int off = 8; off >= 1; off >>= 1) {
      ps[j]  += __shfl_xor(ps[j],  off, 16);
      ps2[j] += __shfl_xor(ps2[j], off, 16);
    }
  }
  if ((lane & 15) == 0) {
#pragma unroll
    for (int j = 0; j < 8; ++j) {
      int row = rg * 16 + mbase + j;
      atomicAdd(&s1[row], ps[j]);
      atomicAdd(&s2[row], ps2[j]);
    }
  }
}

// ---------------------------------------------------------------------------
// multi = [a0*x | a1*x] @ Cbig^T  (single K=2560 GEMM, alpha folded into A);
// x = LN(multi + residue) -> xf, xbf
// ---------------------------------------------------------------------------
__global__ __launch_bounds__(512, 1) void k_multi(const bfu* __restrict__ Xbf,
                                                  const bfu* __restrict__ Cbig,
                                                  const float* __restrict__ att,
                                                  const int* __restrict__ seg,
                                                  const float* __restrict__ mx,
                                                  const float* __restrict__ dn,
                                                  const float* __restrict__ res,
                                                  const float* __restrict__ gam,
                                                  const float* __restrict__ bet,
                                                  float* __restrict__ xf,
                                                  bfu* __restrict__ xbf)
{
  extern __shared__ char smem[];
  bfu*   Abig  = (bfu*)smem;                          // 32 x 2560 bf16 (160 KB)
  float* alpha = (float*)(smem + 32 * DIM2 * 2);      // 64
  float* s1    = alpha + 64;                          // 32
  float* s2    = s1 + 32;                             // 32
  float* muv   = s2 + 32;                             // 64
  const int tid = threadIdx.x;
  const long r0 = (long)blockIdx.x * 32;
  if (tid < 32) {
    s1[tid] = 0.f; s2[tid] = 0.f;
    int g = seg[r0 + tid];
    for (int h = 0; h < NHEAD; ++h)
      alpha[tid * 2 + h] =
          __expf(att[(r0 + tid) * 2 + h] - mx[g * 2 + h]) / dn[g * 2 + h];
  }
  __syncthreads();
  // Build Abig = [alpha0 .* x , alpha1 .* x], one global read per x element
  for (int i = tid; i < 32 * DIM / 8; i += 512) {
    int idx = i * 8;
    int row = idx / DIM, d = idx - row * DIM;
    v8us xv = *(const v8us*)(Xbf + (r0 + row) * DIM + d);
    float a0 = alpha[row * 2], a1 = alpha[row * 2 + 1];
    v8us o0, o1;
#pragma unroll
    for (int e = 0; e < 8; ++e) {
      float x = bf2f(xv[e]);
      o0[e] = f2bf(a0 * x);
      o1[e] = f2bf(a1 * x);
    }
    *(v8us*)(Abig + (long)row * DIM2 + d)       = o0;
    *(v8us*)(Abig + (long)row * DIM2 + DIM + d) = o1;
  }
  __syncthreads();
  const int wave = tid >> 5, lane = tid & 31;
  const int rg = wave >> 3, cg = wave & 7;
  const int mbase = (lane < 16) ? 0 : 8;
  v8f acc[10] = {};
  wmma_panel<10>(Abig + rg * 16 * DIM2, DIM2, DIM2 / 32, Cbig, DIM2, cg, lane, acc);
  // add residual, then in-register LayerNorm
#pragma unroll
  for (int t = 0; t < 10; ++t) {
    int nn = ((cg + (t << 3)) << 4) + (lane & 15);
#pragma unroll
    for (int j = 0; j < 8; ++j)
      acc[t][j] += res[(r0 + rg * 16 + mbase + j) * DIM + nn];
  }
  ln_accumulate(acc, 10, s1, s2, rg, mbase, lane);
  __syncthreads();
  if (tid < 32) {
    float m   = s1[tid] * (1.f / DIM);
    float var = fmaxf(s2[tid] * (1.f / DIM) - m * m, 0.f);
    muv[tid * 2] = m; muv[tid * 2 + 1] = rsqrtf(var + LN_EPS);
  }
  __syncthreads();
#pragma unroll
  for (int t = 0; t < 10; ++t) {
    int nn = ((cg + (t << 3)) << 4) + (lane & 15);
    float gv = gam[nn], bv = bet[nn];
#pragma unroll
    for (int j = 0; j < 8; ++j) {
      int row = rg * 16 + mbase + j;
      float v = (acc[t][j] - muv[row * 2]) * muv[row * 2 + 1] * gv + bv;
      long idx = (r0 + row) * DIM + nn;
      xf[idx]  = v;
      xbf[idx] = f2bf(v);
    }
  }
}

// ---------------------------------------------------------------------------
// Fused FFN: h1 = relu(x W1^T + b1) lives in LDS (32x2560 bf16, 160 KB);
// GEMM1 runs as two NT=10 column halves (keeps bu[]+acc under budget);
// out = LN(h1 W2^T + b2 + x), LayerNorm done in-register.
// ---------------------------------------------------------------------------
__global__ __launch_bounds__(512, 1) void k_ffn(const bfu* __restrict__ xbf,
                                                const float* __restrict__ xf,
                                                const bfu* __restrict__ W1b,
                                                const float* __restrict__ b1,
                                                const bfu* __restrict__ W2b,
                                                const float* __restrict__ b2,
                                                const float* __restrict__ gam,
                                                const float* __restrict__ bet,
                                                float* __restrict__ out)
{
  extern __shared__ char smem[];
  bfu*   Ap  = (bfu*)smem;                               // 32 x 1280 bf16 (80 KB)
  bfu*   H1  = (bfu*)(smem + 32 * DIM * 2);              // 32 x 2560 bf16 (160 KB)
  float* s1  = (float*)(smem + 32 * DIM * 2 + 32 * DIM2 * 2);  // 32
  float* s2  = s1 + 32;                                  // 32
  float* muv = s2 + 32;                                  // 64
  const int tid = threadIdx.x;
  const long r0 = (long)blockIdx.x * 32;
  async_panel_load(Ap, xbf + r0 * DIM, 32 * DIM * 2, tid, 512);
  if (tid < 32) { s1[tid] = 0.f; s2[tid] = 0.f; }
  __syncthreads();
  const int wave = tid >> 5, lane = tid & 31;
  const int rg = wave >> 3, cg = wave & 7;
  const int mbase = (lane < 16) ? 0 : 8;
  // GEMM1: [32,1280] x [1280,2560] -> relu -> H1 (LDS), two column halves
  for (int half = 0; half < 2; ++half) {
    v8f acc[10] = {};
    wmma_panel<10>(Ap + rg * 16 * DIM, DIM, DIM / 32,
                   W1b + (long)half * DIM * DIM, DIM, cg, lane, acc);
#pragma unroll
    for (int t = 0; t < 10; ++t) {
      int nn = half * DIM + ((cg + (t << 3)) << 4) + (lane & 15);
      float bias = b1[nn];
#pragma unroll
      for (int j = 0; j < 8; ++j) {
        float v = acc[t][j] + bias;
        H1[(rg * 16 + mbase + j) * DIM2 + nn] = f2bf(v > 0.f ? v : 0.f);
      }
    }
  }
  __syncthreads();
  {   // GEMM2: [32,2560] x [2560,1280] + b2 + x -> in-register LN -> out
    v8f acc[10] = {};
    wmma_panel<10>(H1 + rg * 16 * DIM2, DIM2, DIM2 / 32, W2b, DIM2, cg, lane, acc);
#pragma unroll
    for (int t = 0; t < 10; ++t) {
      int nn = ((cg + (t << 3)) << 4) + (lane & 15);
      float bias = b2[nn];
#pragma unroll
      for (int j = 0; j < 8; ++j)
        acc[t][j] += bias + xf[(r0 + rg * 16 + mbase + j) * DIM + nn];
    }
    ln_accumulate(acc, 10, s1, s2, rg, mbase, lane);
    __syncthreads();
    if (tid < 32) {
      float m   = s1[tid] * (1.f / DIM);
      float var = fmaxf(s2[tid] * (1.f / DIM) - m * m, 0.f);
      muv[tid * 2] = m; muv[tid * 2 + 1] = rsqrtf(var + LN_EPS);
    }
    __syncthreads();
#pragma unroll
    for (int t = 0; t < 10; ++t) {
      int nn = ((cg + (t << 3)) << 4) + (lane & 15);
      float gv = gam[nn], bv = bet[nn];
#pragma unroll
      for (int j = 0; j < 8; ++j) {
        int row = rg * 16 + mbase + j;
        float v = (acc[t][j] - muv[row * 2]) * muv[row * 2 + 1] * gv + bv;
        out[(r0 + row) * DIM + nn] = v;
      }
    }
  }
}

// ---------------------------------------------------------------------------
extern "C" void kernel_launch(void* const* d_in, const int* in_sizes, int n_in,
                              void* d_out, int out_size, void* d_ws, size_t ws_size,
                              hipStream_t stream) {
  (void)in_sizes; (void)n_in; (void)out_size; (void)ws_size;
  const float* residue = (const float*)d_in[0];
  const int*   seg     = (const int*)d_in[1];
  const float* Wq      = (const float*)d_in[2];
  const float* Wk      = (const float*)d_in[3];
  const float* Wv      = (const float*)d_in[4];
  const float* Wc      = (const float*)d_in[5];
  const float* W1      = (const float*)d_in[6];
  const float* b1      = (const float*)d_in[7];
  const float* W2      = (const float*)d_in[8];
  const float* b2      = (const float*)d_in[9];
  const float* ln_g    = (const float*)d_in[10];
  const float* ln_b    = (const float*)d_in[11];

  char* ws = (char*)d_ws;
  bfu*   Xbf  = (bfu*)(ws + 0L);            // 167,772,160 B
  bfu*   xbf  = (bfu*)(ws + 167772160L);    // 167,772,160 B
  float* xf   = (float*)(ws + 335544320L);  // 335,544,320 B
  float* att  = (float*)(ws + 671088640L);  //     524,288 B
  float* mxb  = (float*)(ws + 671612928L);  //       8,192 B
  float* dnb  = (float*)(ws + 671621120L);  //       8,192 B
  bfu*   Mt   = (bfu*)(ws + 671629312L);    //   6,553,600 B
  bfu*   Cbig = (bfu*)(ws + 678182912L);    //   6,553,600 B
  bfu*   W1b  = (bfu*)(ws + 684736512L);    //   6,553,600 B
  bfu*   W2b  = (bfu*)(ws + 691290112L);    //   6,553,600 B

  const long nX = (long)N_NODES * DIM;
  const long nW = (long)DIM2 * DIM;
  k_tobf<<<(unsigned)((nX / 4 + 255) / 256), 256, 0, stream>>>(residue, Xbf, nX);
  k_tobf<<<(unsigned)((nW / 4 + 255) / 256), 256, 0, stream>>>(W1, W1b, nW);
  k_tobf<<<(unsigned)((nW / 4 + 255) / 256), 256, 0, stream>>>(W2, W2b, nW);

  dim3 gP(DIM / 16, DIM / 16, NHEAD);
  k_mt<<<gP, 256, 0, stream>>>(Wq, Wk, Mt);
  k_ct<<<gP, 256, 0, stream>>>(Wv, Wc, Cbig);

  const size_t shAtt = 32 * DIM * 2 + 64 * 4;                 // 82,176 B
  k_att<<<N_NODES / 32, 512, shAtt, stream>>>(Xbf, Mt, att);
  k_segstats<<<NGRAPH, 256, 0, stream>>>(att, seg, mxb, dnb);

  const size_t shMulti = 32 * DIM2 * 2 + (64 + 32 + 32 + 64) * 4;  // 164,608 B
  k_multi<<<N_NODES / 32, 512, shMulti, stream>>>(Xbf, Cbig, att, seg, mxb, dnb,
                                                  residue, ln_g, ln_b, xf, xbf);

  const size_t shFfn = 32 * DIM * 2 + 32 * DIM2 * 2 + (32 + 32 + 64) * 4;  // 246,272 B
  k_ffn<<<N_NODES / 32, 512, shFfn, stream>>>(xbf, xf, W1b, b1, W2b, b2,
                                              ln_g, ln_b, (float*)d_out);
}
